// RGCN_45200235823220
// MI455X (gfx1250) — matrix operand order repaired and checked by reading, not compiled, not run on hardware
//
#include <hip/hip_runtime.h>
#include <hip/hip_bf16.h>

typedef _Float16 half_t;
typedef __attribute__((ext_vector_type(16))) _Float16 v16h;
typedef __attribute__((ext_vector_type(8)))  float    v8f;

#define N_NODE 50000
#define N_RELS 4
#define DIM    128
#define KCAT   640   // 4*128 (relation blocks) + 128 (root)
#define NEG_SLOPE 0.01f

// ---------------------------------------------------------------- utilities
__global__ void zero_f32(float* p, long n) {
    long i = blockIdx.x * (long)blockDim.x + threadIdx.x;
    if (i < n) p[i] = 0.0f;
}

__global__ void gather_rows(const float* __restrict__ emb, const int* __restrict__ ids,
                            float* __restrict__ out, long total) {
    long i = blockIdx.x * (long)blockDim.x + threadIdx.x;
    if (i >= total) return;
    long n = i / DIM, c = i % DIM;
    out[i] = emb[(size_t)ids[n] * DIM + c];
}

// ------------------------------------------------- edge scatter (1 wave/edge)
__global__ void scatter_edges(const float* __restrict__ x,
                              const int* __restrict__ src, const int* __restrict__ dst,
                              const int* __restrict__ typ,
                              float* __restrict__ s, float* __restrict__ cnt, int E) {
    long idx = blockIdx.x * (long)blockDim.x + threadIdx.x;
    int e = (int)(idx >> 5);
    int lane = (int)(idx & 31);
    if (e >= E) return;
    int sn = src[e], dn = dst[e], t = typ[e];
    size_t seg = (size_t)dn * N_RELS + t;
    const float4 v = *(const float4*)(x + (size_t)sn * DIM + lane * 4);
    float* sp = s + seg * DIM + lane * 4;
    atomicAdd(sp + 0, v.x);
    atomicAdd(sp + 1, v.y);
    atomicAdd(sp + 2, v.z);
    atomicAdd(sp + 3, v.w);
    if (lane == 0) atomicAdd(cnt + seg, 1.0f);
}

// ------------------- combined weight: rows 0..511 = comp@basis, 512..639 = root
// stored TRANSPOSED f16: WcatT[j*640 + k]  (j = out col, k = reduction index)
__global__ void make_wcat(const float* __restrict__ comp, const float* __restrict__ basis,
                          const float* __restrict__ root, half_t* __restrict__ WcatT) {
    int id = blockIdx.x * blockDim.x + threadIdx.x;      // 128*640
    if (id >= DIM * KCAT) return;
    int j = id / KCAT, k = id % KCAT;
    float v;
    if (k < N_RELS * DIM) {
        int r = k >> 7, i = k & 127;
        float acc = 0.f;
        for (int b = 0; b < 16; ++b)
            acc += comp[r * 16 + b] * basis[((size_t)(b * DIM) + i) * DIM + j];
        v = acc;
    } else {
        v = root[(size_t)(k - N_RELS * DIM) * DIM + j];
    }
    WcatT[(size_t)j * KCAT + k] = (half_t)v;
}

// ---------------- A matrix: [n, 640] f16 = [mean-normalized s (512) | x (128)]
__global__ void build_a16(const float* __restrict__ s, const float* __restrict__ cnt,
                          const float* __restrict__ x, half_t* __restrict__ A16) {
    long id = blockIdx.x * (long)blockDim.x + threadIdx.x;  // N_NODE*640
    if (id >= (long)N_NODE * KCAT) return;
    long n = id / KCAT;
    int  k = (int)(id % KCAT);
    float v;
    if (k < N_RELS * DIM) {
        int r = k >> 7, i = k & 127;
        float c = cnt[n * N_RELS + r];
        v = s[((size_t)n * N_RELS + r) * DIM + i] / fmaxf(c, 1.0f);
    } else {
        v = x[(size_t)n * DIM + (k - N_RELS * DIM)];
    }
    A16[id] = (half_t)v;
}

// --------------------------- generic fp32 weight [K,N] -> f16 transposed [N,K]
__global__ void transpose_to_f16(const float* __restrict__ w, half_t* __restrict__ BT,
                                 int K, int N) {
    long id = blockIdx.x * (long)blockDim.x + threadIdx.x;
    if (id >= (long)K * N) return;
    int n = (int)(id / K), k = (int)(id % K);
    BT[(size_t)n * K + k] = (half_t)w[(size_t)k * N + n];
}

// --------------------- gather user/bundle rows into MLP A matrix [P,256] f16
__global__ void gather_pairs(const float* __restrict__ h, const int* __restrict__ users,
                             const int* __restrict__ bundles, half_t* __restrict__ A16,
                             int P) {
    long id = blockIdx.x * (long)blockDim.x + threadIdx.x;  // P*256
    if (id >= (long)P * 2 * DIM) return;
    int p = (int)(id / (2 * DIM));
    int k = (int)(id % (2 * DIM));
    float v = (k < DIM) ? h[(size_t)users[p] * DIM + k]
                        : h[(size_t)bundles[p] * DIM + (k - DIM)];
    A16[id] = (half_t)v;
}

// ------------------------------------------------------------- WMMA GEMM
typedef union { v16h v; unsigned u[8]; } frag_t;

template <int NT>
__device__ __forceinline__ void wmma_k_step(const half_t* __restrict__ Arow,
                                            const half_t* __restrict__ Bbase,
                                            int K, int k0, bool loadNext,
                                            frag_t& a_use, frag_t& a_nxt,
                                            v8f (&acc)[NT]) {
    constexpr int H = NT / 2;
    // first half of B fragments (small live set: H fragments)
    frag_t bl[H];
#pragma unroll
    for (int t = 0; t < H; ++t) {
        const unsigned* bp = (const unsigned*)(Bbase + (size_t)t * 16 * K + k0);
        bl[t].u[0] = bp[0]; bl[t].u[1] = bp[1]; bl[t].u[2] = bp[2]; bl[t].u[3] = bp[3];
        bl[t].u[4] = bp[4]; bl[t].u[5] = bp[5]; bl[t].u[6] = bp[6]; bl[t].u[7] = bp[7];
    }
    // next k-step's A fragment: in flight during both WMMA half-chains
    if (loadNext) {
        const unsigned* ap = (const unsigned*)(Arow + k0 + 32);
        const unsigned* aq = (const unsigned*)(Arow + k0 + 40);
        a_nxt.u[0] = ap[0]; a_nxt.u[1] = ap[1]; a_nxt.u[2] = ap[2]; a_nxt.u[3] = ap[3];
        a_nxt.u[4] = aq[0]; a_nxt.u[5] = aq[1]; a_nxt.u[6] = aq[2]; a_nxt.u[7] = aq[3];
        __builtin_prefetch(Arow + k0 + 64, 0, 3);
    }
#pragma unroll
    for (int t = 0; t < H; ++t)
        acc[t] = __builtin_amdgcn_wmma_f32_16x16x32_f16(
            false, a_use.v, false, bl[t].v, (short)0, acc[t], false, false);
    // second half: loads overlap the first half-chain's WMMA execution
    frag_t bh[NT - H];
#pragma unroll
    for (int t = H; t < NT; ++t) {
        const unsigned* bp = (const unsigned*)(Bbase + (size_t)t * 16 * K + k0);
        bh[t - H].u[0] = bp[0]; bh[t - H].u[1] = bp[1]; bh[t - H].u[2] = bp[2]; bh[t - H].u[3] = bp[3];
        bh[t - H].u[4] = bp[4]; bh[t - H].u[5] = bp[5]; bh[t - H].u[6] = bp[6]; bh[t - H].u[7] = bp[7];
    }
#pragma unroll
    for (int t = H; t < NT; ++t)
        acc[t] = __builtin_amdgcn_wmma_f32_16x16x32_f16(
            false, a_use.v, false, bh[t - H].v, (short)0, acc[t], false, false);
}

// C[M,N] = A[M,K] * B[K,N] (+bias, lrelu).  A: row-major f16.  BT: [N,K] f16.
// One wave computes a 16 x (16*NT) strip; A fragment loaded once per k-step.
// k-loop unrolled by 2 with role-swapped A double buffer (no register copies).
// __launch_bounds__(256,1): allow the full VGPR budget -> no scratch spills.
template <int NT>
__global__ __launch_bounds__(256, 1)
void gemm_wmma(const half_t* __restrict__ A, const half_t* __restrict__ BT,
               const float* __restrict__ bias,
               float* __restrict__ outF32, half_t* __restrict__ outF16,
               int Mtiles, int Ngroups, int Ntiles, int K, int lrelu) {
    int wave = (int)((blockIdx.x * (long)blockDim.x + threadIdx.x) >> 5);
    int lane = threadIdx.x & 31;
    if (wave >= Mtiles * Ngroups) return;         // wave-uniform exit
    int tm = wave / Ngroups;
    int tg = wave % Ngroups;                      // N-tile group (NT tiles each)
    int row = lane & 15;                          // M row (A) / N col (B, C)
    int hi  = lane >> 4;                          // half-wave selector
    int kA = hi * 16;                             // A: K {0..15} / {16..31}
    int kB = hi * 16;                             // B: K {0..15} / {16..31}
    const half_t* Arow  = A  + (size_t)(tm * 16 + row) * K + kA;
    const half_t* Bbase = BT + (size_t)(tg * NT * 16 + row) * K + kB;

    v8f acc[NT];
#pragma unroll
    for (int t = 0; t < NT; ++t) acc[t] = (v8f){};

    // prologue: A fragment for k0 = 0
    frag_t aA, aB;
    {
        const unsigned* ap = (const unsigned*)(Arow);
        const unsigned* aq = (const unsigned*)(Arow + 8);
        aA.u[0] = ap[0]; aA.u[1] = ap[1]; aA.u[2] = ap[2]; aA.u[3] = ap[3];
        aA.u[4] = aq[0]; aA.u[5] = aq[1]; aA.u[6] = aq[2]; aA.u[7] = aq[3];
    }

    int k0 = 0;
    for (; k0 + 64 <= K; k0 += 64) {              // K/32 even at every call site
        wmma_k_step<NT>(Arow, Bbase, K, k0,      true,           aA, aB, acc);
        wmma_k_step<NT>(Arow, Bbase, K, k0 + 32, (k0 + 64 < K),  aB, aA, acc);
    }
    if (k0 < K)                                    // generic tail (unused here)
        wmma_k_step<NT>(Arow, Bbase, K, k0, false, aA, aB, acc);

    // C/D layout: lane<16 -> N=lane, M=v ; lane>=16 -> N=lane-16, M=8+v
    int col   = lane & 15;
    int rbase = hi * 8;
    int Nld   = Ntiles * 16;
#pragma unroll
    for (int t = 0; t < NT; ++t) {
        int tn = tg * NT + t;
        float bval = bias[tn * 16 + col];
#pragma unroll
        for (int v = 0; v < 8; ++v) {
            float val = acc[t][v] + bval;
            if (lrelu) val = (val > 0.f) ? val : NEG_SLOPE * val;
            size_t off = (size_t)(tm * 16 + rbase + v) * Nld + tn * 16 + col;
            if (outF32) outF32[off] = val;
            if (outF16) outF16[off] = (half_t)val;
        }
    }
}

// ------------------------------------------------------- final 64->1 project
__global__ void final_out(const half_t* __restrict__ a3, const float* __restrict__ w,
                          const float* __restrict__ b, float* __restrict__ out, int P) {
    int p = blockIdx.x * blockDim.x + threadIdx.x;
    if (p >= P) return;
    float s = 0.f;
    for (int j = 0; j < 64; ++j) s += (float)a3[(size_t)p * 64 + j] * w[j];
    out[p] = s + b[0];
}

// =========================================================================
static inline char* ws_take(char*& cur, size_t bytes) {
    char* p = cur;
    cur += (bytes + 255) & ~(size_t)255;
    return p;
}

extern "C" void kernel_launch(void* const* d_in, const int* in_sizes, int n_in,
                              void* d_out, int out_size, void* d_ws, size_t ws_size,
                              hipStream_t stream) {
    const float* embedding = (const float*)d_in[0];
    const float* comp1  = (const float*)d_in[1];
    const float* basis1 = (const float*)d_in[2];
    const float* root1  = (const float*)d_in[3];
    const float* bias1  = (const float*)d_in[4];
    const float* comp2  = (const float*)d_in[5];
    const float* basis2 = (const float*)d_in[6];
    const float* root2  = (const float*)d_in[7];
    const float* bias2  = (const float*)d_in[8];
    const float* fc1_w  = (const float*)d_in[9];
    const float* fc1_b  = (const float*)d_in[10];
    const float* fc2_w  = (const float*)d_in[11];
    const float* fc2_b  = (const float*)d_in[12];
    const float* fc3_w  = (const float*)d_in[13];
    const float* fc3_b  = (const float*)d_in[14];
    const float* out_w  = (const float*)d_in[15];
    const float* out_b  = (const float*)d_in[16];
    const int* x_ids    = (const int*)d_in[17];
    const int* edge_src = (const int*)d_in[18];
    const int* edge_dst = (const int*)d_in[19];
    const int* edge_typ = (const int*)d_in[20];
    const int* users    = (const int*)d_in[21];
    const int* bundles  = (const int*)d_in[22];
    const int E = in_sizes[18];
    const int P = in_sizes[21];

    char* cur = (char*)d_ws;
    float*  s_buf = (float*)ws_take(cur, (size_t)N_NODE * N_RELS * DIM * 4);
    float*  c_buf = (float*)ws_take(cur, (size_t)N_NODE * N_RELS * 4);
    half_t* A16   = (half_t*)ws_take(cur, (size_t)N_NODE * KCAT * 2);
    half_t* WcatT = (half_t*)ws_take(cur, (size_t)DIM * KCAT * 2);
    float*  h0    = (float*)ws_take(cur, (size_t)N_NODE * DIM * 4);
    float*  h1    = (float*)ws_take(cur, (size_t)N_NODE * DIM * 4);
    float*  h2    = (float*)ws_take(cur, (size_t)N_NODE * DIM * 4);
    half_t* mlpA  = (half_t*)ws_take(cur, (size_t)P * 2 * DIM * 2);
    half_t* w1T   = (half_t*)ws_take(cur, (size_t)256 * 256 * 2);
    half_t* w2T   = (half_t*)ws_take(cur, (size_t)128 * 256 * 2);
    half_t* w3T   = (half_t*)ws_take(cur, (size_t)64 * 128 * 2);
    half_t* a1    = (half_t*)ws_take(cur, (size_t)P * 256 * 2);
    half_t* a2    = (half_t*)ws_take(cur, (size_t)P * 128 * 2);
    half_t* a3    = (half_t*)ws_take(cur, (size_t)P * 64 * 2);

    const int TB = 256;
    auto blocks = [](long n, int tb) { return (unsigned)((n + tb - 1) / tb); };

    // x = embedding[x_ids]
    long nh = (long)N_NODE * DIM;
    gather_rows<<<blocks(nh, TB), TB, 0, stream>>>(embedding, x_ids, h0, nh);

    const float* xs[2]    = { h0, h1 };
    float*       hs[2]    = { h1, h2 };
    const float* comps[2] = { comp1, comp2 };
    const float* bass[2]  = { basis1, basis2 };
    const float* roots[2] = { root1, root2 };
    const float* biases[2]= { bias1, bias2 };

    for (int l = 0; l < 2; ++l) {
        long ns = (long)N_NODE * N_RELS * DIM;
        zero_f32<<<blocks(ns, TB), TB, 0, stream>>>(s_buf, ns);
        zero_f32<<<blocks((long)N_NODE * N_RELS, TB), TB, 0, stream>>>(c_buf, (long)N_NODE * N_RELS);
        scatter_edges<<<blocks((long)E * 32, TB), TB, 0, stream>>>(
            xs[l], edge_src, edge_dst, edge_typ, s_buf, c_buf, E);
        make_wcat<<<blocks((long)DIM * KCAT, TB), TB, 0, stream>>>(
            comps[l], bass[l], roots[l], WcatT);
        build_a16<<<blocks((long)N_NODE * KCAT, TB), TB, 0, stream>>>(
            s_buf, c_buf, xs[l], A16);
        // [50000 x 640] @ [640 x 128] : one wave per 16x128 strip -> 3125 waves
        int Mt = N_NODE / 16;
        gemm_wmma<8><<<blocks((long)Mt * 32, TB), TB, 0, stream>>>(
            A16, WcatT, biases[l], hs[l], nullptr, Mt, 1, DIM / 16, KCAT, 1);
    }

    // MLP head on pairs
    gather_pairs<<<blocks((long)P * 2 * DIM, TB), TB, 0, stream>>>(h2, users, bundles, mlpA, P);
    transpose_to_f16<<<blocks((long)256 * 256, TB), TB, 0, stream>>>(fc1_w, w1T, 256, 256);
    transpose_to_f16<<<blocks((long)256 * 128, TB), TB, 0, stream>>>(fc2_w, w2T, 256, 128);
    transpose_to_f16<<<blocks((long)128 * 64, TB), TB, 0, stream>>>(fc3_w, w3T, 128, 64);

    int Pt = P / 16;
    // fc1: [P,256]@[256,256] -> two groups of 8 N-tiles per M row
    gemm_wmma<8><<<blocks((long)Pt * 2 * 32, TB), TB, 0, stream>>>(
        mlpA, w1T, fc1_b, nullptr, a1, Pt, 2, 16, 256, 1);
    // fc2: [P,256]@[256,128] -> one group of 8 N-tiles
    gemm_wmma<8><<<blocks((long)Pt * 32, TB), TB, 0, stream>>>(
        a1, w2T, fc2_b, nullptr, a2, Pt, 1, 8, 256, 1);
    // fc3: [P,128]@[128,64] -> one group of 4 N-tiles
    gemm_wmma<4><<<blocks((long)Pt * 32, TB), TB, 0, stream>>>(
        a2, w3T, fc3_b, nullptr, a3, Pt, 1, 4, 128, 1);

    final_out<<<blocks(P, TB), TB, 0, stream>>>(a3, out_w, out_b, (float*)d_out, P);
}
